// PCNAttentionExploration_85246510891348
// MI455X (gfx1250) — matrix operand order repaired and checked by reading, not compiled, not run on hardware
//
#include <hip/hip_runtime.h>

typedef __attribute__((ext_vector_type(16))) __bf16 v16bf;
typedef __attribute__((ext_vector_type(8)))  float  v8f;
typedef __attribute__((ext_vector_type(4)))  int    v4i;

#define B_    2
#define T_    1024
#define H_    1024
#define NH_   16
#define HD_   64
#define S_    3
#define SLICE (NH_ * T_ * HD_)   /* 1048576 per (s,b) / per b  */
#define TOT   (B_ * SLICE)       /* 2097152                    */
#define ROWS  (B_ * T_)          /* 2048                       */

#if __has_builtin(__builtin_amdgcn_global_load_async_to_lds_b128)
#define HAVE_ASYNC_LDS 1
#else
#define HAVE_ASYNC_LDS 0
#endif

// ---------------------------------------------------------------------------
// helpers
// ---------------------------------------------------------------------------

// 16-byte global -> LDS copy via GLOBAL_LOAD_ASYNC_TO_LDS_B128 (ASYNCcnt).
__device__ __forceinline__ void copy16_g2l(const __bf16* g, __bf16* l) {
#if HAVE_ASYNC_LDS
  __builtin_amdgcn_global_load_async_to_lds_b128(
      (__attribute__((address_space(1))) v4i*)(unsigned long long)g,
      (__attribute__((address_space(3))) v4i*)(unsigned int)(unsigned long long)l,
      0, 0);
#else
  *(uint4*)(void*)l = *(const uint4*)(const void*)g;
#endif
}

__device__ __forceinline__ void async_wait() {
#if HAVE_ASYNC_LDS
#if __has_builtin(__builtin_amdgcn_s_wait_asynccnt)
  __builtin_amdgcn_s_wait_asynccnt(0);
#else
  asm volatile("s_wait_asynccnt 0" ::: "memory");
#endif
#endif
}

// wave-local LDS ordering (producer lanes -> consumer lanes within one wave)
__device__ __forceinline__ void wave_lds_fence() {
  asm volatile("s_wait_dscnt 0x0" ::: "memory");
}

// A fragment (bf16, 16x32, MxK) gathered from a row-major LDS tile.
// lanes 0-15 hold M=0..15 with K 0..15,
// lanes 16-31 hold M=0..15 with K 16..31 (matches frag_b).
__device__ __forceinline__ v16bf frag_a(const __bf16* t, int ld, int row_base, int k_base) {
  int lane = threadIdx.x & 31;
  const __bf16* rp = t + (row_base + (lane & 15)) * ld + k_base;
  int kh = (lane & 16) ? 16 : 0;
  v16bf a;
#pragma unroll
  for (int e = 0; e < 16; ++e) a[e] = rp[kh + e];
  return a;
}

// B fragment (bf16, 32x16, KxN). element (k,n) = base[n*stride_n + k*stride_k].
__device__ __forceinline__ v16bf frag_b(const __bf16* base, int stride_n, int stride_k,
                                        int n_base, int k_base) {
  int lane = threadIdx.x & 31;
  const __bf16* p = base + (n_base + (lane & 15)) * stride_n
                         + (k_base + ((lane & 16) ? 16 : 0)) * stride_k;
  v16bf b;
#pragma unroll
  for (int e = 0; e < 16; ++e) b[e] = p[e * stride_k];
  return b;
}

#define WMMA_BF16(a, b, c) \
  __builtin_amdgcn_wmma_f32_16x16x32_bf16(false, (a), false, (b), (short)0, (c), false, false)

// ---------------------------------------------------------------------------
// f32 -> bf16 cast (for weights / hidden so GEMM staging is raw async bytes)
// ---------------------------------------------------------------------------
__global__ __launch_bounds__(256)
void cast_bf16_kernel(const float* __restrict__ src, __bf16* __restrict__ dst, int n) {
  int i = blockIdx.x * 256 + threadIdx.x;
  if (i < n) dst[i] = (__bf16)src[i];
}

// ---------------------------------------------------------------------------
// GEMM: C[M,N] = A[M,K] @ B[N,K]^T + bias ; A,B bf16, f32 accum.
// 128 threads (4 waves), 64x64 tile, K-chunks of 64, async double-buffered LDS.
// Fragment loads are batched ahead of the WMMA chain so DS waits are partial.
// ---------------------------------------------------------------------------
__device__ __forceinline__ void gemm_stage_tiles(const __bf16* __restrict__ A,
                                                 const __bf16* __restrict__ Bw,
                                                 int K, int row0, int col0, int kc,
                                                 __bf16* As, __bf16* Bs, int tid) {
  for (int c = tid; c < 512; c += 128) {       // 64x64 bf16 = 512 x 16B chunks
    int r = c >> 3, c8 = (c & 7) << 3;
    copy16_g2l(A  + (size_t)(row0 + r) * K + kc + c8, As + r * 72 + c8);
    copy16_g2l(Bw + (size_t)(col0 + r) * K + kc + c8, Bs + r * 72 + c8);
  }
}

__global__ __launch_bounds__(128)
void gemm_bt_kernel(const __bf16* __restrict__ A, const __bf16* __restrict__ Bw,
                    const float* __restrict__ bias, float* __restrict__ C,
                    int M, int N, int K) {
  constexpr int LD = 72;
  __shared__ __bf16 As[2][64 * LD];
  __shared__ __bf16 Bs[2][64 * LD];

  int nTilesN = N >> 6;
  int tn = blockIdx.x % nTilesN;
  int tm = blockIdx.x / nTilesN;
  int row0 = tm << 6, col0 = tn << 6;
  int tid = threadIdx.x, wave = tid >> 5, lane = tid & 31;

  v8f acc[4] = {};

  int nK = K >> 6;
  gemm_stage_tiles(A, Bw, K, row0, col0, 0, As[0], Bs[0], tid);

  for (int t = 0; t < nK; ++t) {
    int cur = t & 1;
    async_wait();
    __syncthreads();
    if (t + 1 < nK)
      gemm_stage_tiles(A, Bw, K, row0, col0, (t + 1) << 6, As[1 - cur], Bs[1 - cur], tid);

    // batch all fragment loads, then the WMMA chain
    v16bf a0 = frag_a(As[cur], LD, wave * 16, 0);
    v16bf a1 = frag_a(As[cur], LD, wave * 16, 32);
    v16bf b0[4], b1[4];
#pragma unroll
    for (int j = 0; j < 4; ++j) {
      b0[j] = frag_b(Bs[cur], LD, 1, j * 16, 0);    // B(k,n)=Bs[n*LD+k]
      b1[j] = frag_b(Bs[cur], LD, 1, j * 16, 32);
    }
#pragma unroll
    for (int j = 0; j < 4; ++j) acc[j] = WMMA_BF16(a0, b0[j], acc[j]);
#pragma unroll
    for (int j = 0; j < 4; ++j) acc[j] = WMMA_BF16(a1, b1[j], acc[j]);
  }

  int n = lane & 15;
  int mb = (lane & 16) ? 8 : 0;
#pragma unroll
  for (int j = 0; j < 4; ++j) {
    int col = col0 + j * 16 + n;
    float bv = bias[col];
#pragma unroll
    for (int r = 0; r < 8; ++r) {
      int row = row0 + wave * 16 + mb + r;
      C[(size_t)row * N + col] = acc[j][r] + bv;
    }
  }
}

// ---------------------------------------------------------------------------
// Fused LayerNorm + ReLU, writes bf16 for next GEMM. One block per row.
// ---------------------------------------------------------------------------
__global__ __launch_bounds__(256)
void ln_relu_kernel(const float* __restrict__ y, const float* __restrict__ g,
                    const float* __restrict__ bb, __bf16* __restrict__ xb) {
  __shared__ float s0[256], s1[256];
  int row = blockIdx.x, tid = threadIdx.x;
  const float* yr = y + (size_t)row * H_;
  float a = 0.f, b = 0.f;
  for (int i = tid; i < H_; i += 256) { float v = yr[i]; a += v; b += v * v; }
  s0[tid] = a; s1[tid] = b;
  __syncthreads();
  for (int o = 128; o; o >>= 1) {
    if (tid < o) { s0[tid] += s0[tid + o]; s1[tid] += s1[tid + o]; }
    __syncthreads();
  }
  float mu  = s0[0] * (1.0f / H_);
  float var = s1[0] * (1.0f / H_) - mu * mu;
  float inv = rsqrtf(var + 1e-5f);
  for (int i = tid; i < H_; i += 256) {
    float v = (yr[i] - mu) * inv * g[i] + bb[i];
    xb[(size_t)row * H_ + i] = (__bf16)fmaxf(v, 0.f);
  }
}

// ---------------------------------------------------------------------------
// Build q_s (closed form of the 5 PCN steps) + alignment partials.
// ---------------------------------------------------------------------------
__global__ __launch_bounds__(256)
void build_q_kernel(const float* __restrict__ queries, const float* __restrict__ hyp,
                    const float* __restrict__ refn, const float* __restrict__ ctx,
                    const float* __restrict__ keys, float* __restrict__ qf,
                    __bf16* __restrict__ qb, float* __restrict__ pal) {
  __shared__ float sred[256];
  const float a5 = 0.9509900498999999f;   // 0.99^5
  const float cc = 0.0490099501f;          // 0.01 * sum_{i<5} 0.99^i
  const float n0 = 0.0096059601f, n1 = 0.00970299f, n2 = 0.009801f, n3 = 0.0099f;

  int sb = blockIdx.y;
  int s = sb / B_, b = sb % B_;
  size_t qoff = (size_t)sb * SLICE;
  size_t koff = (size_t)b * SLICE;
  size_t r0 = ((size_t)(s * 4 + 0) * B_ + b) * SLICE;
  size_t r1 = ((size_t)(s * 4 + 1) * B_ + b) * SLICE;
  size_t r2 = ((size_t)(s * 4 + 2) * B_ + b) * SLICE;
  size_t r3 = ((size_t)(s * 4 + 3) * B_ + b) * SLICE;

  float asum = 0.f;
  int end = (blockIdx.x + 1) * 4096;
  for (int i = blockIdx.x * 4096 + threadIdx.x; i < end; i += 256) {
    int head = i >> 16, rem = i & 65535, t = rem >> 6, d = rem & 63;
    float cv = ctx[(size_t)(b * T_ + t) * H_ + head * 64 + d];
    float v = a5 * (queries[koff + i] + 0.1f * hyp[qoff + i]) + cc * cv
            + n0 * refn[r0 + i] + n1 * refn[r1 + i]
            + n2 * refn[r2 + i] + n3 * refn[r3 + i];
    qf[qoff + i] = v;
    qb[qoff + i] = (__bf16)v;
    asum += v * keys[koff + i];
  }
  sred[threadIdx.x] = asum;
  __syncthreads();
  for (int o = 128; o; o >>= 1) {
    if (threadIdx.x < o) sred[threadIdx.x] += sred[threadIdx.x + o];
    __syncthreads();
  }
  if (threadIdx.x == 0) pal[sb * 256 + blockIdx.x] = sred[0];
}

// ---------------------------------------------------------------------------
// bf16 cast of K/V + copy keys/values to output slots 1 and 2.
// ---------------------------------------------------------------------------
__global__ __launch_bounds__(256)
void kvcast_kernel(const float* __restrict__ keys, const float* __restrict__ values,
                   __bf16* __restrict__ kbf, __bf16* __restrict__ vbf,
                   float* __restrict__ out) {
  size_t i = (size_t)blockIdx.x * 256 + threadIdx.x;   // < TOT
  float k = keys[i], v = values[i];
  kbf[i] = (__bf16)k;
  vbf[i] = (__bf16)v;
  out[(size_t)TOT + i]     = k;
  out[(size_t)2 * TOT + i] = v;
}

// ---------------------------------------------------------------------------
// Flash attention + energy stats. grid.x = S*B*NH*8 ; 256 threads (8 waves).
// 128 q rows per block; 16 K/V blocks of 64 keys, async double-buffered.
// ---------------------------------------------------------------------------
__global__ __launch_bounds__(256)
void attn_kernel(const __bf16* __restrict__ qb, const __bf16* __restrict__ kbf,
                 const __bf16* __restrict__ vbf, float* __restrict__ pmag,
                 float* __restrict__ pent) {
  constexpr int LD = 72;
  __shared__ __bf16 Qt[128 * LD];
  __shared__ __bf16 Kt[2][64 * LD];
  __shared__ __bf16 Vt[2][64 * LD];
  __shared__ __bf16 Pt[8 * 16 * LD];
  __shared__ float  red[16];

  int bid = blockIdx.x;
  int rb  = bid & 7;
  int sbh = bid >> 3;
  int h   = sbh % NH_;
  int sb  = sbh / NH_;
  int b   = sb % B_;

  size_t qoff  = (((size_t)sb * NH_ + h) * T_ + rb * 128) * HD_;
  size_t kvoff = (((size_t)b  * NH_ + h) * T_) * HD_;

  int tid = threadIdx.x, wave = tid >> 5, lane = tid & 31;

  // stage Q strip (128x64) and K/V block 0 via async-to-LDS
  for (int c = tid; c < 1024; c += 256) {
    int r = c >> 3, c8 = (c & 7) << 3;
    copy16_g2l(qb + qoff + c * 8, Qt + r * LD + c8);
  }
  for (int c = tid; c < 512; c += 256) {
    int r = c >> 3, c8 = (c & 7) << 3;
    copy16_g2l(kbf + kvoff + c * 8, Kt[0] + r * LD + c8);
    copy16_g2l(vbf + kvoff + c * 8, Vt[0] + r * LD + c8);
  }
  async_wait();
  __syncthreads();

  v16bf qa0 = frag_a(Qt, LD, wave * 16, 0);
  v16bf qa1 = frag_a(Qt, LD, wave * 16, 32);

  v8f   O[4] = {};
  float mR[8], Zr[8], Ea[8];
#pragma unroll
  for (int r = 0; r < 8; ++r) { mR[r] = -1e30f; Zr[r] = 0.f; Ea[r] = 0.f; }

  __bf16* Pw = Pt + wave * 16 * LD;

  for (int kb = 0; kb < 16; ++kb) {
    int cur = kb & 1;
    if (kb) { async_wait(); __syncthreads(); }
    if (kb + 1 < 16) {   // prefetch next K/V block into the other buffer
      size_t ko = kvoff + (size_t)(kb + 1) * 64 * 64;
      for (int c = tid; c < 512; c += 256) {
        int r = c >> 3, c8 = (c & 7) << 3;
        copy16_g2l(kbf + ko + c * 8, Kt[1 - cur] + r * LD + c8);
        copy16_g2l(vbf + ko + c * 8, Vt[1 - cur] + r * LD + c8);
      }
    }

    // S = q K^T / sqrt(64): batch 8 K-fragment loads, then 8 WMMAs
    v16bf bk0[4], bk1[4];
#pragma unroll
    for (int j = 0; j < 4; ++j) {
      bk0[j] = frag_b(Kt[cur], LD, 1, j * 16, 0);
      bk1[j] = frag_b(Kt[cur], LD, 1, j * 16, 32);
    }
    v8f c8v[4] = {};
#pragma unroll
    for (int j = 0; j < 4; ++j) c8v[j] = WMMA_BF16(qa0, bk0[j], c8v[j]);
#pragma unroll
    for (int j = 0; j < 4; ++j) c8v[j] = WMMA_BF16(qa1, bk1[j], c8v[j]);

    float lg[4][8];
#pragma unroll
    for (int j = 0; j < 4; ++j)
#pragma unroll
      for (int r = 0; r < 8; ++r) lg[j][r] = c8v[j][r] * 0.125f;

    // online softmax row stats (rows split per 16-lane half)
    float rmax[8];
#pragma unroll
    for (int r = 0; r < 8; ++r) {
      float m = lg[0][r];
      m = fmaxf(m, lg[1][r]); m = fmaxf(m, lg[2][r]); m = fmaxf(m, lg[3][r]);
#pragma unroll
      for (int msk = 1; msk <= 8; msk <<= 1) m = fmaxf(m, __shfl_xor(m, msk));
      rmax[r] = m;
    }
    float alpha[8], pv[4][8], sZ[8], sA[8];
#pragma unroll
    for (int r = 0; r < 8; ++r) {
      float mN = fmaxf(mR[r], rmax[r]);
      alpha[r] = __expf(mR[r] - mN);
      mR[r] = mN;
      sZ[r] = 0.f; sA[r] = 0.f;
    }
#pragma unroll
    for (int j = 0; j < 4; ++j)
#pragma unroll
      for (int r = 0; r < 8; ++r) {
        float p = __expf(lg[j][r] - mR[r]);
        pv[j][r] = p;
        sZ[r] += p;
        sA[r] += p * lg[j][r];
      }
#pragma unroll
    for (int r = 0; r < 8; ++r) {
#pragma unroll
      for (int msk = 1; msk <= 8; msk <<= 1) {
        sZ[r] += __shfl_xor(sZ[r], msk);
        sA[r] += __shfl_xor(sA[r], msk);
      }
      Zr[r] = Zr[r] * alpha[r] + sZ[r];
      Ea[r] = Ea[r] * alpha[r] + sA[r];
    }
#pragma unroll
    for (int j = 0; j < 4; ++j)
#pragma unroll
      for (int r = 0; r < 8; ++r) O[j][r] *= alpha[r];

    // P round-trips through per-wave LDS scratch (C-frag -> A-frag relayout).
    // Per-wave private area: wave-local s_wait_dscnt suffices, no barrier.
    int n = lane & 15, mb = (lane & 16) ? 8 : 0;
#pragma unroll
    for (int j = 0; j < 4; ++j)
#pragma unroll
      for (int r = 0; r < 8; ++r)
        Pw[(mb + r) * LD + j * 16 + n] = (__bf16)pv[j][r];
    wave_lds_fence();

    // batch P and V fragment loads, then 8 WMMAs
    v16bf pa0 = frag_a(Pw, LD, 0, 0);
    v16bf pa1 = frag_a(Pw, LD, 0, 32);
    v16bf bv0[4], bv1[4];
#pragma unroll
    for (int jj = 0; jj < 4; ++jj) {
      bv0[jj] = frag_b(Vt[cur], 1, LD, jj * 16, 0);
      bv1[jj] = frag_b(Vt[cur], 1, LD, jj * 16, 32);
    }
#pragma unroll
    for (int jj = 0; jj < 4; ++jj) O[jj] = WMMA_BF16(pa0, bv0[jj], O[jj]);
#pragma unroll
    for (int jj = 0; jj < 4; ++jj) O[jj] = WMMA_BF16(pa1, bv1[jj], O[jj]);
  }

  // per-row: ||out||, entropy = M + log Z - Ea/Z
  float magsum = 0.f, entsum = 0.f;
#pragma unroll
  for (int r = 0; r < 8; ++r) {
    float inv = 1.f / Zr[r];
    float m2 = 0.f;
#pragma unroll
    for (int jj = 0; jj < 4; ++jj) { float o = O[jj][r] * inv; m2 += o * o; }
#pragma unroll
    for (int msk = 1; msk <= 8; msk <<= 1) m2 += __shfl_xor(m2, msk);
    float ent = mR[r] + __logf(Zr[r]) - Ea[r] * inv;
    if ((lane & 15) == 0) { magsum += sqrtf(m2); entsum += ent; }
  }
  magsum += __shfl_xor(magsum, 16);
  entsum += __shfl_xor(entsum, 16);
  if (lane == 0) { red[wave] = magsum; red[8 + wave] = entsum; }
  __syncthreads();
  if (tid == 0) {
    float mg = 0.f, en = 0.f;
#pragma unroll
    for (int w = 0; w < 8; ++w) { mg += red[w]; en += red[8 + w]; }
    int pidx = sb * (NH_ * 8) + h * 8 + rb;
    pmag[pidx] = mg;
    pent[pidx] = en;
  }
}

// ---------------------------------------------------------------------------
// Energy reduction: one block per (s,b).
// ---------------------------------------------------------------------------
__global__ __launch_bounds__(128)
void energy_kernel(const float* __restrict__ pmag, const float* __restrict__ pent,
                   const float* __restrict__ pal, float* __restrict__ E) {
  __shared__ float sm[128], se[128], sa[128];
  int sb = blockIdx.x, tid = threadIdx.x;
  sm[tid] = pmag[sb * 128 + tid];
  se[tid] = pent[sb * 128 + tid];
  sa[tid] = pal[sb * 256 + tid] + pal[sb * 256 + 128 + tid];
  __syncthreads();
  for (int o = 64; o; o >>= 1) {
    if (tid < o) { sm[tid] += sm[tid + o]; se[tid] += se[tid + o]; sa[tid] += sa[tid + o]; }
    __syncthreads();
  }
  if (tid == 0) {
    const float inv = 1.0f / (NH_ * T_);
    E[sb] = (sm[0] - 0.1f * se[0] - 0.05f * sa[0]) * inv;
  }
}

// ---------------------------------------------------------------------------
// Final: probs = softmax(-E) over S; selected = sum_s p * q_s.
// ---------------------------------------------------------------------------
__global__ __launch_bounds__(256)
void finalize_kernel(const float* __restrict__ qf, const float* __restrict__ E,
                     float* __restrict__ out) {
  size_t idx = (size_t)blockIdx.x * 256 + threadIdx.x;   // < TOT
  int b = (int)(idx >> 20);                              // slice per b is 1M
  float e0 = -E[0 * B_ + b], e1 = -E[1 * B_ + b], e2 = -E[2 * B_ + b];
  float m = fmaxf(e0, fmaxf(e1, e2));
  float w0 = __expf(e0 - m), w1 = __expf(e1 - m), w2 = __expf(e2 - m);
  float inv = 1.f / (w0 + w1 + w2);
  out[idx] = inv * (w0 * qf[idx] + w1 * qf[idx + (size_t)TOT] + w2 * qf[idx + (size_t)2 * TOT]);
}

// ---------------------------------------------------------------------------
extern "C" void kernel_launch(void* const* d_in, const int* in_sizes, int n_in,
                              void* d_out, int out_size, void* d_ws, size_t ws_size,
                              hipStream_t stream) {
  (void)in_sizes; (void)n_in; (void)out_size; (void)ws_size;
  const float* queries = (const float*)d_in[0];
  const float* keys    = (const float*)d_in[1];
  const float* values  = (const float*)d_in[2];
  const float* hs      = (const float*)d_in[3];
  const float* W1      = (const float*)d_in[4];
  const float* b1      = (const float*)d_in[5];
  const float* lng     = (const float*)d_in[6];
  const float* lnb     = (const float*)d_in[7];
  const float* W2      = (const float*)d_in[8];
  const float* b2      = (const float*)d_in[9];
  const float* hyp     = (const float*)d_in[10];
  const float* refn    = (const float*)d_in[11];
  float* out = (float*)d_out;

  char* w = (char*)d_ws;
  float*  y    = (float*)(w + 0);          // 2048*1024 f32   (8 MB)
  __bf16* xb   = (__bf16*)(w + 8388608);   // 2048*1024 bf16  (4 MB)
  float*  ctx  = (float*)(w + 12582912);   // 2048*1024 f32   (8 MB)
  float*  qf   = (float*)(w + 20971520);   // 6M f32          (24 MB)
  __bf16* qb   = (__bf16*)(w + 46137344);  // 6M bf16         (12 MB)
  __bf16* kbf  = (__bf16*)(w + 58720256);  // 2M bf16         (4 MB)
  __bf16* vbf  = (__bf16*)(w + 62914560);  // 2M bf16         (4 MB)
  __bf16* hsb  = (__bf16*)(w + 67108864);  // 2M bf16         (4 MB)
  __bf16* W1b  = (__bf16*)(w + 71303168);  // 1M bf16         (2 MB)
  __bf16* W2b  = (__bf16*)(w + 73400320);  // 1M bf16         (2 MB)
  float*  pal  = (float*)(w + 75497472);   // 6*256 f32
  float*  pmag = (float*)(w + 75503616);   // 6*128 f32
  float*  pent = (float*)(w + 75506688);   // 6*128 f32
  float*  E    = (float*)(w + 75509760);   // 6 f32

  // 0) bf16 casts so GEMM staging is raw async-to-LDS bytes
  cast_bf16_kernel<<<(ROWS * H_) / 256, 256, 0, stream>>>(hs, hsb, ROWS * H_);
  cast_bf16_kernel<<<(H_ * H_) / 256, 256, 0, stream>>>(W1, W1b, H_ * H_);
  cast_bf16_kernel<<<(H_ * H_) / 256, 256, 0, stream>>>(W2, W2b, H_ * H_);
  // 1) y = hs @ W1^T + b1
  gemm_bt_kernel<<<(ROWS / 64) * (H_ / 64), 128, 0, stream>>>(hsb, W1b, b1, y, ROWS, H_, H_);
  // 2) LayerNorm + ReLU -> bf16
  ln_relu_kernel<<<ROWS, 256, 0, stream>>>(y, lng, lnb, xb);
  // 3) ctx = x @ W2^T + b2
  gemm_bt_kernel<<<(ROWS / 64) * (H_ / 64), 128, 0, stream>>>(xb, W2b, b2, ctx, ROWS, H_, H_);
  // 4) q_s closed form + alignment partials
  build_q_kernel<<<dim3(256, S_ * B_), 256, 0, stream>>>(queries, hyp, refn, ctx, keys, qf, qb, pal);
  // 5) bf16 K/V + copy keys/values to output
  kvcast_kernel<<<TOT / 256, 256, 0, stream>>>(keys, values, kbf, vbf, out);
  // 6) flash attention + energy stats
  attn_kernel<<<S_ * B_ * NH_ * 8, 256, 0, stream>>>(qb, kbf, vbf, pmag, pent);
  // 7) energies
  energy_kernel<<<S_ * B_, 128, 0, stream>>>(pmag, pent, pal, E);
  // 8) softmax-weighted mix -> selected
  finalize_kernel<<<TOT / 256, 256, 0, stream>>>(qf, E, out);
}
